// IntraAttention_Decoder_48902497632706
// MI455X (gfx1250) — matrix operand order repaired and checked by reading, not compiled, not run on hardware
//
#include <hip/hip_runtime.h>
#include <hip/hip_bf16.h>
#include <math.h>

typedef __attribute__((ext_vector_type(16))) __bf16 v16bf;
typedef __attribute__((ext_vector_type(8)))  float  v8f;

#define BATCH 32
#define TLEN  2048
#define HDIM  1024
#define APAD  1032   // 1024 + 8 halfs padding -> LDS row stride 2064B (516 dwords, %64==4): conflict-free
#define MTILE 64     // t-rows per workgroup: 4x B reuse (across waves) -> W_prev L2 traffic 2GB total

__device__ __forceinline__ unsigned short f2bf(float f) {
    unsigned u = __float_as_uint(f);
    u += 0x7FFFu + ((u >> 16) & 1u);          // round-to-nearest-even
    return (unsigned short)(u >> 16);
}

// fast tanh: 1 - 2/(e^{2x}+1); exact at +/-inf, ~1e-7 rel error, ~4 ops vs ~20 for libm
__device__ __forceinline__ float tanh_fast(float x) {
    float e = __expf(2.0f * x);
    return 1.0f - 2.0f / (e + 1.0f);
}

// ---------------- K0: W_prev f32 -> bf16 (2MB, lives in L2, reused by every WG) ------------
__global__ void cvt_wprev_kernel(const float* __restrict__ W, unsigned short* __restrict__ Wb) {
    int i = blockIdx.x * blockDim.x + threadIdx.x;          // over H*H/4 float4s
    float4 w = ((const float4*)W)[i];
    ushort2 lo; lo.x = f2bf(w.x); lo.y = f2bf(w.y);
    ushort2 hi; hi.x = f2bf(w.z); hi.y = f2bf(w.w);
    ((ushort2*)Wb)[2 * i]     = lo;
    ((ushort2*)Wb)[2 * i + 1] = hi;
}

// ---------------- K1: dec_fea[b,d] = s_t[b,:] . W_s[d,:] + b_s[d] --------------------------
__global__ void dec_fea_kernel(const float* __restrict__ s_t, const float* __restrict__ W_s,
                               const float* __restrict__ b_s, float* __restrict__ dec) {
    int idx = blockIdx.x * blockDim.x + threadIdx.x;        // B*H threads
    int b = idx >> 10, d = idx & (HDIM - 1);
    const float4* sr = (const float4*)(s_t + (size_t)b * HDIM);
    const float4* wr = (const float4*)(W_s + (size_t)d * HDIM);
    float acc = 0.f;
    for (int j = 0; j < HDIM / 4; ++j) {
        float4 s = sr[j], w = wr[j];
        acc += s.x * w.x + s.y * w.y + s.z * w.z + s.w * w.w;
    }
    dec[idx] = acc + b_s[d];
}

// ---------------- K2: fused  scores = tanh(prev_s @ W_prev^T + dec) . v  (WMMA bf16) -------
// One WG = 64 t-rows x full H; A-panel (64x1024 bf16, 129KB) staged once in dynamic LDS.
// Wave w owns M-subtile (w&3) and n-tile group (w>>2): round-1's proven single-stream shape
// per wave (compiler hoists the wave's 32 A-frags into high VGPRs), while the 4 waves of a
// group fetch identical B-tiles near-simultaneously -> WGP$ dedups -> 2MB unique L2/WG.
union Frag { uint4 q[2]; v16bf v; };

#define LOADB(dst, k0) do {                                          \
    (dst).q[0] = *(const uint4*)(browp + (k0));                      \
    (dst).q[1] = *(const uint4*)(browp + (k0) + 16); } while (0)
#define LOADA(dst, ap, k0) do {                                      \
    (dst).q[0] = *(const uint4*)((ap) + (k0));                       \
    (dst).q[1] = *(const uint4*)((ap) + (k0) + 16); } while (0)
#define WMMA_BF16(acc, fa, fb)                                       \
    (acc) = __builtin_amdgcn_wmma_f32_16x16x32_bf16(false, (fa).v, false, (fb).v, \
                                                    (short)0, (acc), false, false)

__global__ __launch_bounds__(256)
void attn_scores_wmma(const float* __restrict__ prev_s,
                      const unsigned short* __restrict__ Wb,
                      const float* __restrict__ dec,
                      const float* __restrict__ vvec,
                      float* __restrict__ scores) {
    extern __shared__ __align__(16) unsigned char smem_raw[];
    unsigned short* Apanel = (unsigned short*)smem_raw;                 // [MTILE][APAD] bf16
    float* sScore = (float*)(smem_raw + (size_t)MTILE * APAD * 2);      // [MTILE]

    const int b   = blockIdx.y;
    const int t0  = blockIdx.x * MTILE;
    const int tid = threadIdx.x;

    // Stage + convert A panel: thread tid owns cols 4*tid..4*tid+3 of every row (coalesced)
    {
        const float4* src = (const float4*)(prev_s + ((size_t)b * TLEN + t0) * HDIM);
        for (int j = 0; j < MTILE; ++j) {
            float4 x = src[(size_t)j * (HDIM / 4) + tid];
            unsigned short* dst = &Apanel[j * APAD + 4 * tid];
            dst[0] = f2bf(x.x); dst[1] = f2bf(x.y); dst[2] = f2bf(x.z); dst[3] = f2bf(x.w);
        }
    }
    if (tid < MTILE) sScore[tid] = 0.f;
    __syncthreads();

    const int lane  = tid & 31;
    const int wave  = tid >> 5;
    const int lmod  = lane & 15;   // A: row M (within subtile) / B,C: col N
    const int lhalf = lane >> 4;   // K-half selector per documented 16-bit layouts
    const int mi    = wave & 3;    // M-subtile owned by this wave (16 t-rows)
    const int grp   = wave >> 2;   // n-tile group: nt = grp, grp+2, ...

    // A-frag base for this wave's subtile: runs K {base..+7} and {base+16..+23}, base=lhalf*8
    const unsigned short* aBase = &Apanel[(mi * 16 + lmod) * APAD + lhalf * 8];

    float sc[8];
    #pragma unroll
    for (int r = 0; r < 8; ++r) sc[r] = 0.f;

    for (int nt = grp; nt < HDIM / 16; nt += 2) {
        const int n0 = nt * 16;
        v8f acc = {};
        // B-frag: W_prev_bf16 row d = n0+lmod; lanes 0-15 carry K 0..7,16..23, lanes 16-31 K 8..15,24..31
        const unsigned short* browp = Wb + (size_t)(n0 + lmod) * HDIM + lhalf * 8;

        for (int kt = 0; kt < HDIM / 32; ++kt) {
            const int k0 = kt * 32;
            Frag fa, fb;
            LOADA(fa, aBase, k0);                     // 2x ds_load_b128 (nt-invariant: hoistable)
            LOADB(fb, k0);                            // 2x global_load_b128 (L2-resident W)
            WMMA_BF16(acc, fa, fb);
        }

        // Fused epilogue: tanh + v projection entirely in registers (et never hits memory)
        const float dv = dec[b * HDIM + n0 + lmod];
        const float vv = vvec[n0 + lmod];
        #pragma unroll
        for (int r = 0; r < 8; ++r)
            sc[r] += tanh_fast(acc[r] + dv) * vv;     // D[M=r+8*lhalf][N=lmod]
    }

    // Reduce over N: sum across the 16 lanes of each half, then merge waves via LDS atomics
    #pragma unroll
    for (int r = 0; r < 8; ++r) {
        float x = sc[r];
        x += __shfl_xor(x, 1, 16);
        x += __shfl_xor(x, 2, 16);
        x += __shfl_xor(x, 4, 16);
        x += __shfl_xor(x, 8, 16);
        sc[r] = x;
    }
    if (lmod == 0) {
        #pragma unroll
        for (int r = 0; r < 8; ++r)
            atomicAdd(&sScore[mi * 16 + lhalf * 8 + r], sc[r]);   // ds_add_f32 (2 waves/row)
    }
    __syncthreads();
    if (tid < MTILE) scores[(size_t)b * TLEN + t0 + tid] = sScore[tid];
}

// ---------------- K3: softmax over T + context ct = at . prev_s ----------------------------
__global__ __launch_bounds__(256)
void softmax_ctx_kernel(const float* __restrict__ scores, const float* __restrict__ prev_s,
                        float* __restrict__ ct) {
    __shared__ float sAt[TLEN];
    __shared__ float red[32];
    const int b = blockIdx.x, tid = threadIdx.x;

    float m = -1e30f;
    for (int t = tid; t < TLEN; t += 256) {
        float s = scores[(size_t)b * TLEN + t];
        sAt[t] = s; m = fmaxf(m, s);
    }
    for (int off = 16; off > 0; off >>= 1) m = fmaxf(m, __shfl_xor(m, off, 32));
    if ((tid & 31) == 0) red[tid >> 5] = m;
    __syncthreads();
    if (tid == 0) {
        float mm = red[0];
        for (int i = 1; i < 8; ++i) mm = fmaxf(mm, red[i]);
        red[8] = mm;
    }
    __syncthreads();
    const float mm = red[8];

    float sum = 0.f;
    for (int t = tid; t < TLEN; t += 256) {
        float e = __expf(sAt[t] - mm);
        sAt[t] = e; sum += e;
    }
    for (int off = 16; off > 0; off >>= 1) sum += __shfl_xor(sum, off, 32);
    if ((tid & 31) == 0) red[16 + (tid >> 5)] = sum;
    __syncthreads();
    if (tid == 0) {
        float s = 0.f;
        for (int i = 0; i < 8; ++i) s += red[16 + i];
        red[9] = 1.f / s;
    }
    __syncthreads();
    const float inv = red[9];

    float acc[4] = {0.f, 0.f, 0.f, 0.f};
    const float* base = prev_s + (size_t)b * TLEN * HDIM;
    for (int t = 0; t < TLEN; ++t) {
        const float w = sAt[t];
        const float* row = base + (size_t)t * HDIM;
        #pragma unroll
        for (int j = 0; j < 4; ++j) acc[j] += w * row[tid + 256 * j];   // coalesced columns
    }
    #pragma unroll
    for (int j = 0; j < 4; ++j) ct[(size_t)b * HDIM + tid + 256 * j] = acc[j] * inv;
}

// ---------------- K4: prev_s_new = concat(prev_s, s_t) (float4 streaming copy) -------------
__global__ void build_prev_new_kernel(const float* __restrict__ prev_s,
                                      const float* __restrict__ s_t,
                                      float* __restrict__ out) {
    long long g = (long long)blockIdx.x * blockDim.x + threadIdx.x;  // B*2049*256 float4s
    int c = (int)(g & 255);
    long long row = g >> 8;
    int b = (int)(row / (TLEN + 1));
    int t = (int)(row % (TLEN + 1));
    const float4* src = (t < TLEN)
        ? (const float4*)(prev_s + ((size_t)b * TLEN + t) * HDIM) + c
        : (const float4*)(s_t + (size_t)b * HDIM) + c;
    ((float4*)out)[row * 256 + c] = *src;
}

extern "C" void kernel_launch(void* const* d_in, const int* in_sizes, int n_in,
                              void* d_out, int out_size, void* d_ws, size_t ws_size,
                              hipStream_t stream) {
    const float* s_t    = (const float*)d_in[0];
    const float* prev_s = (const float*)d_in[1];
    const float* W_prev = (const float*)d_in[2];
    const float* W_s    = (const float*)d_in[3];
    const float* b_s    = (const float*)d_in[4];
    const float* v      = (const float*)d_in[5];
    float* out = (float*)d_out;                       // [ct_d (B*H)] ++ [prev_s_new (B*(T+1)*H)]

    // workspace layout
    unsigned short* Wb  = (unsigned short*)d_ws;                       // 2 MB bf16 W_prev
    float* dec          = (float*)((char*)d_ws + 2u * 1024u * 1024u);  // 128 KB
    float* scores       = dec + BATCH * HDIM;                          // 256 KB

    const size_t smem_bytes = (size_t)MTILE * APAD * 2 + MTILE * sizeof(float);  // ~129.3 KB dyn LDS

    cvt_wprev_kernel<<<(HDIM * HDIM / 4) / 256, 256, 0, stream>>>(W_prev, Wb);
    dec_fea_kernel<<<(BATCH * HDIM) / 256, 256, 0, stream>>>(s_t, W_s, b_s, dec);
    attn_scores_wmma<<<dim3(TLEN / MTILE, BATCH), 256, smem_bytes, stream>>>(prev_s, Wb, dec, v, scores);
    softmax_ctx_kernel<<<BATCH, 256, 0, stream>>>(scores, prev_s, out);
    build_prev_new_kernel<<<BATCH * (TLEN + 1), 256, 0, stream>>>(prev_s, s_t, out + BATCH * HDIM);
}
